// ExpertAdaRMSLayer_4123168604687
// MI455X (gfx1250) — matrix operand order repaired and checked by reading, not compiled, not run on hardware
//
#include <hip/hip_runtime.h>
#include <hip/hip_bf16.h>
#include <math.h>

// ---- types ----
typedef __bf16 bf16_t;
typedef __attribute__((ext_vector_type(16))) __bf16 v16bf;
typedef __attribute__((ext_vector_type(8)))  __bf16 v8bf;
typedef __attribute__((ext_vector_type(8)))  float  v8f;

#define HIDDEN 2048
#define NQ     8
#define NKV    1
#define HD     256
#define INTER  8192
#define BB     4
#define SS     2048
#define ROWS   (BB * SS)        // 8192
#define SM_SCALE (1.0f / 16.0f) // 1/sqrt(256)

// CDNA5 async DMA: global -> LDS, 16B per lane, tracked by ASYNCcnt.
__device__ __forceinline__ void async_ld16(const bf16_t* g, bf16_t* l)
{
    unsigned lds = (unsigned)(uintptr_t)l;   // low 32 bits = LDS byte offset
    asm volatile("global_load_async_to_lds_b128 %0, %1, off"
                 :: "v"(lds), "v"(g) : "memory");
}
#define WAIT_ASYNC(n) asm volatile("s_wait_asynccnt " #n ::: "memory")

// =====================================================================
// Small helper kernels
// =====================================================================

// s[b][j] = dot(time_emb[b], tw[j,:])   (tw row-major HIDDENxHIDDEN)
__global__ __launch_bounds__(256) void scale_kernel(
    const float* __restrict__ TE, const float* __restrict__ TW,
    float* __restrict__ S)
{
    int j = blockIdx.x * 256 + threadIdx.x;
    int b = blockIdx.y;
    const float4* te = (const float4*)(TE + (size_t)b * HIDDEN);
    const float4* tw = (const float4*)(TW + (size_t)j * HIDDEN);
    float acc = 0.f;
    for (int k = 0; k < HIDDEN / 4; ++k) {
        float4 a = te[k], w = tw[k];
        acc += a.x * w.x + a.y * w.y + a.z * w.z + a.w * w.w;
    }
    S[(size_t)b * HIDDEN + j] = acc;
}

// AdaRMS: h[row,:] = x*rsqrt(mean(x^2)+eps) * (w * (1+sc[b,:]))  -> bf16
__global__ __launch_bounds__(256) void adarms_kernel(
    const float* __restrict__ X, const float* __restrict__ W,
    const float* __restrict__ Sc, bf16_t* __restrict__ H)
{
    const int row = blockIdx.x;          // 0..ROWS-1
    const int b   = row >> 11;           // row / SS
    const float* x = X + (size_t)row * HIDDEN;
    float ss = 0.f;
    for (int j = threadIdx.x; j < HIDDEN; j += 256) { float v = x[j]; ss += v * v; }
    #pragma unroll
    for (int off = 1; off < 32; off <<= 1) ss += __shfl_xor(ss, off, 32);
    __shared__ float sred[8];
    if ((threadIdx.x & 31) == 0) sred[threadIdx.x >> 5] = ss;
    __syncthreads();
    float tot = 0.f;
    #pragma unroll
    for (int w = 0; w < 8; ++w) tot += sred[w];
    float rs = rsqrtf(tot / (float)HIDDEN + 1e-6f);
    const float* sc = Sc + (size_t)b * HIDDEN;
    for (int j = threadIdx.x; j < HIDDEN; j += 256)
        H[(size_t)row * HIDDEN + j] = (bf16_t)(x[j] * rs * (W[j] * (1.f + sc[j])));
}

__global__ __launch_bounds__(256) void cast_bf16_kernel(
    const float* __restrict__ src, bf16_t* __restrict__ dst, int n)
{
    int i = blockIdx.x * 256 + threadIdx.x;
    if (i < n) dst[i] = (bf16_t)src[i];
}

// RoPE on Q (layout: qkv[row, h*HD + d], output qb[(row*NQ+h)*HD + d])
__global__ __launch_bounds__(256) void rope_q_kernel(
    const float* __restrict__ QKV, const int* __restrict__ pos,
    bf16_t* __restrict__ Qb)
{
    int i = blockIdx.x * 256 + threadIdx.x;     // ROWS*NQ*HD
    int d  = i & 255;
    int h  = (i >> 8) & 7;
    int rs = i >> 11;
    const float* q = QKV + (size_t)rs * 2560 + h * HD;
    float p  = (float)pos[rs];
    int dd   = d & 127;
    float f  = __powf(10000.f, -(float)(2 * dd) / (float)HD);
    float ang = p * f, c = __cosf(ang), s = __sinf(ang);
    float v   = q[d];
    float rot = (d < 128) ? -q[d + 128] : q[d - 128];
    Qb[i] = (bf16_t)(v * c + rot * s);
}

// RoPE on K + cast V  (kv head = 1)
__global__ __launch_bounds__(256) void rope_kv_kernel(
    const float* __restrict__ QKV, const int* __restrict__ pos,
    bf16_t* __restrict__ Kb, bf16_t* __restrict__ Vb)
{
    int i = blockIdx.x * 256 + threadIdx.x;     // ROWS*HD
    int d  = i & 255;
    int rs = i >> 8;
    const float* base = QKV + (size_t)rs * 2560;
    float p  = (float)pos[rs];
    int dd   = d & 127;
    float f  = __powf(10000.f, -(float)(2 * dd) / (float)HD);
    float ang = p * f, c = __cosf(ang), s = __sinf(ang);
    float kv  = base[2048 + d];
    float rot = (d < 128) ? -base[2048 + d + 128] : base[2048 + d - 128];
    Kb[i] = (bf16_t)(kv * c + rot * s);
    Vb[i] = (bf16_t)base[2304 + d];
}

// silu(gate) * up ; GU packed [row, 0..8191]=gate, [row, 8192..]=up
__global__ __launch_bounds__(256) void silu_mul_kernel(
    const bf16_t* __restrict__ GU, bf16_t* __restrict__ Mo)
{
    int i   = blockIdx.x * 256 + threadIdx.x;   // ROWS*INTER
    int col = i & (INTER - 1);
    int row = i >> 13;
    float g = (float)GU[(size_t)row * (2 * INTER) + col];
    float u = (float)GU[(size_t)row * (2 * INTER) + INTER + col];
    float sg = g / (1.f + __expf(-g));
    Mo[i] = (bf16_t)(sg * u);
}

// =====================================================================
// Tiled WMMA GEMM:  C[M,N] = A[M,K](bf16) * B[N,K](bf16)^T  (+ R)
// Block tile 128x256, BK=32, 256 threads = 8 waves (2x4 wave grid).
// Each wave: 64x64 => 4x4 wmma 16x16 subtiles, f32 accumulation.
// Double-buffered LDS fed by global_load_async_to_lds_b128 (ASYNCcnt).
// mode 0: f32 out   mode 1: f32 out + residual R   mode 2: bf16 out
// =====================================================================
__global__ __launch_bounds__(256) void gemm_bf16_wmma(
    const bf16_t* __restrict__ A, const bf16_t* __restrict__ B,
    void* __restrict__ Cout, const float* __restrict__ R,
    int M, int N, int K, int mode)
{
    __shared__ bf16_t As[2][128][40];   // +8 pad: 80B row stride, 16B aligned
    __shared__ bf16_t Bs[2][256][40];

    const int tid  = threadIdx.x;
    const int lane = tid & 31;
    const int wave = tid >> 5;
    const int waveM = wave >> 2;                 // 0..1
    const int waveN = wave & 3;                  // 0..3
    const int bm = blockIdx.y, bn = blockIdx.x;
    const int rowBase = bm * 128 + waveM * 64;
    const int colBase = bn * 256 + waveN * 64;

    v8f c[4][4];
    #pragma unroll
    for (int i = 0; i < 4; ++i)
        #pragma unroll
        for (int j = 0; j < 4; ++j)
            #pragma unroll
            for (int e = 0; e < 8; ++e) c[i][j][e] = 0.f;

    const int fm   = lane & 15;
    const int koff = (lane >> 4) * 8;    // A-frag K sub-chunk base (ISA 16-bit A layout)
    const int kb   = (lane >> 4) * 8;    // B-frag K base (ISA 16-bit B layout)

    // stage one 128x32 A slab + 256x32 B slab into LDS buffer `buf`
    // (6 async b128 per thread: 2 for A, 4 for B)
    auto stage = [&](int buf, int k0) {
        #pragma unroll
        for (int u = 0; u < 2; ++u) {
            int cch = tid * 2 + u;           // 0..511
            int row = cch >> 2;
            int col = (cch & 3) * 8;
            async_ld16(A + (size_t)(bm * 128 + row) * K + k0 + col,
                       &As[buf][row][col]);
        }
        #pragma unroll
        for (int u = 0; u < 4; ++u) {
            int cch = tid * 4 + u;           // 0..1023
            int row = cch >> 2;
            int col = (cch & 3) * 8;
            async_ld16(B + (size_t)(bn * 256 + row) * K + k0 + col,
                       &Bs[buf][row][col]);
        }
    };

    const int nk = K / 32;
    stage(0, 0);
    for (int kt = 0; kt < nk; ++kt) {
        const int buf = kt & 1;
        if (kt + 1 < nk) {
            stage(buf ^ 1, (kt + 1) * 32);
            WAIT_ASYNC(0x6);                 // loads retire in order: current slab done
        } else {
            WAIT_ASYNC(0x0);
        }
        __syncthreads();

        v16bf af[4];
        #pragma unroll
        for (int i = 0; i < 4; ++i) {
            const int r = waveM * 64 + i * 16 + fm;
            v8bf lo = *(const v8bf*)&As[buf][r][koff];
            v8bf hi = *(const v8bf*)&As[buf][r][koff + 16];
            #pragma unroll
            for (int e = 0; e < 8; ++e) { af[i][e] = lo[e]; af[i][8 + e] = hi[e]; }
        }
        #pragma unroll
        for (int j = 0; j < 4; ++j) {
            const int n = waveN * 64 + j * 16 + fm;
            v8bf lo = *(const v8bf*)&Bs[buf][n][kb];
            v8bf hi = *(const v8bf*)&Bs[buf][n][kb + 16];
            v16bf bfr;
            #pragma unroll
            for (int e = 0; e < 8; ++e) { bfr[e] = lo[e]; bfr[8 + e] = hi[e]; }
            #pragma unroll
            for (int i = 0; i < 4; ++i)
                c[i][j] = __builtin_amdgcn_wmma_f32_16x16x32_bf16(
                    false, af[i], false, bfr, (short)0, c[i][j], false, false);
        }
        __syncthreads();   // protect buf before next stage overwrites it
    }

    // Epilogue (C layout: vgpr r, lanes 0-15 -> M=r, lanes 16-31 -> M=r+8)
    const int half = lane >> 4;
    #pragma unroll
    for (int i = 0; i < 4; ++i)
        #pragma unroll
        for (int j = 0; j < 4; ++j)
            #pragma unroll
            for (int r = 0; r < 8; ++r) {
                size_t row = (size_t)(rowBase + i * 16 + r + 8 * half);
                size_t col = (size_t)(colBase + j * 16 + fm);
                size_t idx = row * (size_t)N + col;
                float v = c[i][j][r];
                if (mode == 1) v += R[idx];
                if (mode == 2) ((bf16_t*)Cout)[idx] = (bf16_t)v;
                else           ((float*)Cout)[idx]  = v;
            }
}

// =====================================================================
// Flash attention (full softmax, no mask). grid=(S/64, NQ, B), 128 thr.
// Each wave: 16 queries x HD=256; kv tiles of 32 keys streamed via LDS
// using async-to-LDS DMA. QK^T and P*V both via v_wmma.
// =====================================================================
__global__ __launch_bounds__(128) void attn_wmma_kernel(
    const bf16_t* __restrict__ Qb, const bf16_t* __restrict__ Kb,
    const bf16_t* __restrict__ Vb, bf16_t* __restrict__ Ctx)
{
    __shared__ bf16_t Ks[32][264];       // 528B row stride (16B aligned)
    __shared__ bf16_t Vs[32][264];
    __shared__ bf16_t Ps[4][16][40];     // per-wave P staging (80B rows)

    const int tid  = threadIdx.x;
    const int lane = tid & 31;
    const int wave = tid >> 5;           // 0..3
    const int head = blockIdx.y;
    const int b    = blockIdx.z;
    const int q0   = blockIdx.x * 64 + wave * 16;
    const int fm   = lane & 15;
    const int half = lane >> 4;

    // Preload Q A-fragments: qf[t] covers hd [32t, 32t+32)
    v16bf qf[8];
    const bf16_t* qrow = Qb + ((size_t)(b * SS + q0 + fm) * NQ + head) * HD;
    #pragma unroll
    for (int t = 0; t < 8; ++t) {
        const bf16_t* p = qrow + t * 32 + half * 8;
        v8bf lo = *(const v8bf*)p;
        v8bf hi = *(const v8bf*)(p + 16);
        #pragma unroll
        for (int e = 0; e < 8; ++e) { qf[t][e] = lo[e]; qf[t][8 + e] = hi[e]; }
    }

    v8f acc[16];
    #pragma unroll
    for (int nc = 0; nc < 16; ++nc)
        #pragma unroll
        for (int e = 0; e < 8; ++e) acc[nc][e] = 0.f;
    float m_r[8], l_r[8];
    #pragma unroll
    for (int r = 0; r < 8; ++r) { m_r[r] = -1e30f; l_r[r] = 0.f; }

    const int krow = tid >> 2;           // 0..31
    const int kcol = (tid & 3) * 64;     // 4 slabs of 64 bf16

    for (int kt = 0; kt < SS / 32; ++kt) {
        // Stage K/V tile (32 x 256 bf16 each) via async DMA
        const bf16_t* gk = Kb + (size_t)(b * SS + kt * 32 + krow) * HD + kcol;
        const bf16_t* gv = Vb + (size_t)(b * SS + kt * 32 + krow) * HD + kcol;
        #pragma unroll
        for (int u = 0; u < 8; ++u) {
            async_ld16(gk + u * 8, &Ks[krow][kcol + u * 8]);
            async_ld16(gv + u * 8, &Vs[krow][kcol + u * 8]);
        }
        WAIT_ASYNC(0x0);
        __syncthreads();

        // scores: 16 queries x 32 keys, contraction over hd=256
        v8f sc[2];
        #pragma unroll
        for (int j = 0; j < 2; ++j) {
            #pragma unroll
            for (int e = 0; e < 8; ++e) sc[j][e] = 0.f;
            #pragma unroll
            for (int t = 0; t < 8; ++t) {
                v16bf kfr;   // B-fragment of K^T (hd x kv)
                const bf16_t* p = &Ks[j * 16 + fm][t * 32 + half * 8];
                v8bf lo = *(const v8bf*)p;
                v8bf hi = *(const v8bf*)(p + 16);
                #pragma unroll
                for (int e = 0; e < 8; ++e) { kfr[e] = lo[e]; kfr[8 + e] = hi[e]; }
                sc[j] = __builtin_amdgcn_wmma_f32_16x16x32_bf16(
                    false, qf[t], false, kfr, (short)0, sc[j], false, false);
            }
        }

        // online softmax over this kv tile
        float alpha[8];
        #pragma unroll
        for (int r = 0; r < 8; ++r) {
            float s0 = sc[0][r] * SM_SCALE, s1 = sc[1][r] * SM_SCALE;
            float t = fmaxf(s0, s1);
            #pragma unroll
            for (int off = 1; off < 16; off <<= 1) t = fmaxf(t, __shfl_xor(t, off, 32));
            float mn = fmaxf(m_r[r], t);
            alpha[r] = __expf(m_r[r] - mn);
            m_r[r] = mn;
            float p0 = __expf(s0 - mn), p1 = __expf(s1 - mn);
            sc[0][r] = p0; sc[1][r] = p1;
            float rs = p0 + p1;
            #pragma unroll
            for (int off = 1; off < 16; off <<= 1) rs += __shfl_xor(rs, off, 32);
            l_r[r] = l_r[r] * alpha[r] + rs;
        }
        #pragma unroll
        for (int nc = 0; nc < 16; ++nc)
            #pragma unroll
            for (int r = 0; r < 8; ++r) acc[nc][r] *= alpha[r];

        // Re-layout P: C-layout regs -> LDS -> A-fragment (16x32 bf16)
        #pragma unroll
        for (int r = 0; r < 8; ++r) {
            Ps[wave][r + 8 * half][fm]      = (bf16_t)sc[0][r];
            Ps[wave][r + 8 * half][16 + fm] = (bf16_t)sc[1][r];
        }
        asm volatile("s_wait_dscnt 0" ::: "memory");
        v16bf pf;
        {
            const bf16_t* p = &Ps[wave][fm][half * 8];
            v8bf lo = *(const v8bf*)p;
            v8bf hi = *(const v8bf*)(p + 16);
            #pragma unroll
            for (int e = 0; e < 8; ++e) { pf[e] = lo[e]; pf[8 + e] = hi[e]; }
        }

        // ctx += P (16x32) @ V (32x256): 16 wmma over hd chunks
        #pragma unroll
        for (int nc = 0; nc < 16; ++nc) {
            v16bf vfr;   // B-fragment of V (kv x hd)
            #pragma unroll
            for (int e = 0; e < 16; ++e)
                vfr[e] = Vs[half * 8 + e + (e & 8)][nc * 16 + fm];
            acc[nc] = __builtin_amdgcn_wmma_f32_16x16x32_bf16(
                false, pf, false, vfr, (short)0, acc[nc], false, false);
        }
        __syncthreads();
    }

    // normalize + store ctx (bf16) at [(b,s), head*HD + d]
    #pragma unroll
    for (int nc = 0; nc < 16; ++nc)
        #pragma unroll
        for (int r = 0; r < 8; ++r) {
            float v = acc[nc][r] / l_r[r];
            int qq = q0 + r + 8 * half;
            Ctx[((size_t)(b * SS + qq) * NQ + head) * HD + nc * 16 + fm] = (bf16_t)v;
        }
}

// =====================================================================
// Host-side orchestration
// =====================================================================
extern "C" void kernel_launch(void* const* d_in, const int* in_sizes, int n_in,
                              void* d_out, int out_size, void* d_ws, size_t ws_size,
                              hipStream_t stream)
{
    const float* x      = (const float*)d_in[0];
    const int*   pos    = (const int*)  d_in[1];
    const float* te     = (const float*)d_in[2];
    const float* ln1_w  = (const float*)d_in[3];
    const float* ln1_tw = (const float*)d_in[4];
    const float* ln2_w  = (const float*)d_in[5];
    const float* ln2_tw = (const float*)d_in[6];
    const float* Wq = (const float*)d_in[7];
    const float* Wk = (const float*)d_in[8];
    const float* Wv = (const float*)d_in[9];
    const float* Wo = (const float*)d_in[10];
    const float* Wg = (const float*)d_in[11];
    const float* Wu = (const float*)d_in[12];
    const float* Wd = (const float*)d_in[13];
    float* out = (float*)d_out;

    char* ws = (char*)d_ws;
    size_t off = 0;
    auto alloc = [&](size_t bytes) -> void* {
        void* p = ws + off;
        off = (off + bytes + 255) & ~(size_t)255;
        return p;
    };

    float*  s1    = (float*) alloc((size_t)BB * HIDDEN * 4);
    float*  s2    = (float*) alloc((size_t)BB * HIDDEN * 4);
    bf16_t* h1    = (bf16_t*)alloc((size_t)ROWS * HIDDEN * 2);
    bf16_t* wqkv  = (bf16_t*)alloc((size_t)2560 * HIDDEN * 2);   // Wq|Wk|Wv rows
    bf16_t* wob   = (bf16_t*)alloc((size_t)HIDDEN * HIDDEN * 2);
    bf16_t* wgu   = (bf16_t*)alloc((size_t)(2 * INTER) * HIDDEN * 2); // Wg|Wu rows
    bf16_t* wdb   = (bf16_t*)alloc((size_t)HIDDEN * INTER * 2);
    float*  qkv   = (float*) alloc((size_t)ROWS * 2560 * 4);
    bf16_t* qb    = (bf16_t*)alloc((size_t)ROWS * NQ * HD * 2);
    bf16_t* kb    = (bf16_t*)alloc((size_t)ROWS * HD * 2);
    bf16_t* vb    = (bf16_t*)alloc((size_t)ROWS * HD * 2);
    bf16_t* ctx   = (bf16_t*)alloc((size_t)ROWS * NQ * HD * 2);
    float*  x2    = (float*) alloc((size_t)ROWS * HIDDEN * 4);
    bf16_t* h2    = (bf16_t*)alloc((size_t)ROWS * HIDDEN * 2);
    bf16_t* gu    = (bf16_t*)alloc((size_t)ROWS * (2 * INTER) * 2);
    bf16_t* mbuf  = (bf16_t*)alloc((size_t)ROWS * INTER * 2);

    auto blks = [](long long n) { return dim3((unsigned)((n + 255) / 256)); };

    // weight casts (pack qkv and gate/up)
    cast_bf16_kernel<<<blks(2048LL*2048), 256, 0, stream>>>(Wq, wqkv,                 2048*2048);
    cast_bf16_kernel<<<blks(256LL *2048), 256, 0, stream>>>(Wk, wqkv + 2048LL*2048,   256*2048);
    cast_bf16_kernel<<<blks(256LL *2048), 256, 0, stream>>>(Wv, wqkv + 2304LL*2048,   256*2048);
    cast_bf16_kernel<<<blks(2048LL*2048), 256, 0, stream>>>(Wo, wob,                  2048*2048);
    cast_bf16_kernel<<<blks(8192LL*2048), 256, 0, stream>>>(Wg, wgu,                  8192*2048);
    cast_bf16_kernel<<<blks(8192LL*2048), 256, 0, stream>>>(Wu, wgu + 8192LL*2048,    8192*2048);
    cast_bf16_kernel<<<blks(2048LL*8192), 256, 0, stream>>>(Wd, wdb,                  2048*8192);

    // adaptive scales
    scale_kernel<<<dim3(HIDDEN/256, BB), 256, 0, stream>>>(te, ln1_tw, s1);
    scale_kernel<<<dim3(HIDDEN/256, BB), 256, 0, stream>>>(te, ln2_tw, s2);

    // ada_rms #1 -> h1 (bf16)
    adarms_kernel<<<ROWS, 256, 0, stream>>>(x, ln1_w, s1, h1);

    // QKV projection: [8192 x 2560] = h1 @ wqkv^T
    gemm_bf16_wmma<<<dim3(2560/256, ROWS/128), 256, 0, stream>>>(
        h1, wqkv, qkv, nullptr, ROWS, 2560, HIDDEN, 0);

    // RoPE
    rope_q_kernel <<<blks((long long)ROWS * NQ * HD), 256, 0, stream>>>(qkv, pos, qb);
    rope_kv_kernel<<<blks((long long)ROWS * HD),      256, 0, stream>>>(qkv, pos, kb, vb);

    // attention
    attn_wmma_kernel<<<dim3(SS/64, NQ, BB), 128, 0, stream>>>(qb, kb, vb, ctx);

    // O projection + residual: x2 = x + ctx @ Wo^T
    gemm_bf16_wmma<<<dim3(HIDDEN/256, ROWS/128), 256, 0, stream>>>(
        ctx, wob, x2, x, ROWS, HIDDEN, HIDDEN, 1);

    // ada_rms #2 -> h2 (bf16)
    adarms_kernel<<<ROWS, 256, 0, stream>>>(x2, ln2_w, s2, h2);

    // gate|up: [8192 x 16384] bf16 = h2 @ wgu^T
    gemm_bf16_wmma<<<dim3((2*INTER)/256, ROWS/128), 256, 0, stream>>>(
        h2, wgu, gu, nullptr, ROWS, 2*INTER, HIDDEN, 2);

    // silu(gate)*up -> mbuf (bf16)
    silu_mul_kernel<<<blks((long long)ROWS * INTER), 256, 0, stream>>>(gu, mbuf);

    // down projection + residual: out = x2 + mbuf @ Wd^T
    gemm_bf16_wmma<<<dim3(HIDDEN/256, ROWS/128), 256, 0, stream>>>(
        mbuf, wdb, out, x2, ROWS, HIDDEN, INTER, 1);
}
